// QuanvolutionHybrid_65481071407109
// MI455X (gfx1250) — matrix operand order repaired and checked by reading, not compiled, not run on hardware
//
#include <hip/hip_runtime.h>
#include <hip/hip_bf16.h>
#include <math.h>

// ---------------- problem constants ----------------
#define BATCH 8192
#define PROTO 2048
#define FEAT  784
#define KP    800            // FEAT padded to a multiple of 32 (WMMA bf16 K)
#define MTILE 128            // rows per block (8 waves x 16)
#define NTILE 16             // prototypes per LDS tile
#define NTILES (PROTO / NTILE)          // 128 tiles
#define TILE_B128 (NTILE * KP * 2 / 16) // 1600 16-byte chunks per tile

typedef __attribute__((ext_vector_type(16))) __bf16 v16bf;
typedef __attribute__((ext_vector_type(8)))  __bf16 v8bf;
typedef __attribute__((ext_vector_type(8)))  float  v8f;

// pointer types matching the async-to-LDS builtin signature:
// (v4i addrspace(1)*, v4i addrspace(3)*, imm offset, imm cpol)
typedef int v4i_vs __attribute__((vector_size(16)));
typedef __attribute__((address_space(1))) v4i_vs as1_v4i;  // global
typedef __attribute__((address_space(3))) v4i_vs as3_v4i;  // LDS

__device__ __forceinline__ unsigned short f32_to_bf16_rne(float f) {
    unsigned int u = __float_as_uint(f);
    unsigned int r = (u + 0x7FFFu + ((u >> 16) & 1u)) >> 16;
    return (unsigned short)r;
}

// CDNA5 async DMA: global -> LDS, tracked by ASYNCcnt (no VGPR data path).
__device__ __forceinline__ void async_copy_b128(const void* g, void* l) {
#if __has_builtin(__builtin_amdgcn_global_load_async_to_lds_b128)
    __builtin_amdgcn_global_load_async_to_lds_b128((as1_v4i*)g, (as3_v4i*)l,
                                                   0, 0);
#else
    asm volatile("global_load_async_to_lds_b128 %0, %1, off"
                 :
                 : "v"((as3_v4i*)l), "v"((as1_v4i*)g)
                 : "memory");
#endif
}

__device__ __forceinline__ void wait_async0() {
#if __has_builtin(__builtin_amdgcn_s_wait_asynccnt)
    __builtin_amdgcn_s_wait_asynccnt(0);
#else
    asm volatile("s_wait_asynccnt 0x0" ::: "memory");
#endif
}

// ---------------------------------------------------------------------------
// Kernel 1: 2x2 stride-2 conv (4 out channels) -> feats[b, 800] in bf16,
//           plus fp32 row norm f2[b]. One block per batch row.
// ---------------------------------------------------------------------------
__global__ __launch_bounds__(256)
void quanv_conv_pack(const float* __restrict__ x,
                     const float* __restrict__ cw,
                     const float* __restrict__ cb,
                     unsigned short* __restrict__ featsu,
                     float* __restrict__ f2) {
    __shared__ float red[256];
    const int b   = blockIdx.x;
    const int tid = threadIdx.x;
    const float* xb = x + (size_t)b * 784;   // (1,28,28)

    float s = 0.0f;
    for (int idx = tid; idx < KP; idx += 256) {
        float v = 0.0f;
        if (idx < FEAT) {
            const int c   = idx / 196;          // channel (14*14 = 196)
            const int rem = idx - c * 196;
            const int i   = rem / 14;
            const int j   = rem - i * 14;
            const float* p = xb + (2 * i) * 28 + 2 * j;
            const float* w = cw + c * 4;        // OIHW, I=1
            v = p[0] * w[0] + p[1] * w[1] + p[28] * w[2] + p[29] * w[3] + cb[c];
            s += v * v;
        }
        featsu[(size_t)b * KP + idx] = f32_to_bf16_rne(v);
    }
    red[tid] = s;
    __syncthreads();
    for (int off = 128; off > 0; off >>= 1) {
        if (tid < off) red[tid] += red[tid + off];
        __syncthreads();
    }
    if (tid == 0) f2[b] = red[0];
}

// ---------------------------------------------------------------------------
// Kernel 2: prototypes fp32 -> bf16 (K padded to 800) + fp32 norms p2[p].
// ---------------------------------------------------------------------------
__global__ __launch_bounds__(256)
void proto_pack(const float* __restrict__ pr,
                unsigned short* __restrict__ protou,
                float* __restrict__ p2) {
    __shared__ float red[256];
    const int p   = blockIdx.x;
    const int tid = threadIdx.x;

    float s = 0.0f;
    for (int idx = tid; idx < KP; idx += 256) {
        float v = 0.0f;
        if (idx < FEAT) {
            v = pr[(size_t)p * FEAT + idx];
            s += v * v;
        }
        protou[(size_t)p * KP + idx] = f32_to_bf16_rne(v);
    }
    red[tid] = s;
    __syncthreads();
    for (int off = 128; off > 0; off >>= 1) {
        if (tid < off) red[tid] += red[tid + off];
        __syncthreads();
    }
    if (tid == 0) p2[p] = red[0];
}

// ---------------------------------------------------------------------------
// Kernel 3: fused bf16 WMMA GEMM (cross = feats @ protos^T) + RBF kernel
//           + linear head + sigmoid + log_softmax.
//   - 64 blocks x 256 threads; each wave owns a 16-row A stripe that the
//     compiler keeps register-resident across the whole prototype sweep.
//   - B tiles (16 protos x 800 K, 25.6 KB) are DOUBLE-BUFFERED in LDS and
//     staged with GLOBAL_LOAD_ASYNC_TO_LDS_B128 (ASYNCcnt), overlapping the
//     DMA of tile nt+1 with the 25 WMMAs + exp epilogue of tile nt.
//   - Kernel matrix never leaves registers: per-row weighted-exp partials
//     are reduced with shfl_xor at the end.
// ---------------------------------------------------------------------------
__global__ __launch_bounds__(256)
void rbf_gemm_head(const unsigned short* __restrict__ featsu,
                   const unsigned short* __restrict__ protou,
                   const float* __restrict__ f2,
                   const float* __restrict__ p2,
                   const float* __restrict__ lin_w,
                   const float* __restrict__ lin_b,
                   float* __restrict__ out) {
    __shared__ __align__(32) unsigned short sB[2][NTILE * KP];  // 2 x 25.6 KB

    const int tid  = threadIdx.x;
    const int wave = tid >> 5;
    const int lane = tid & 31;
    const int nl   = lane & 15;   // N index within 16x16 tile
    const int rg   = lane >> 4;   // row-group / K-group select

    const int m_base = blockIdx.x * MTILE + wave * 16;
    const __bf16* aRow =
        reinterpret_cast<const __bf16*>(featsu) + (size_t)(m_base + nl) * KP;

    // f2 for the 8 rows this lane's accumulator VGPRs map to (M = r + 8*rg)
    float f2r[8];
#pragma unroll
    for (int r = 0; r < 8; ++r) f2r[r] = f2[m_base + rg * 8 + r];

    float rowsum[8] = {0.f, 0.f, 0.f, 0.f, 0.f, 0.f, 0.f, 0.f};

    // ---- prologue: async-stage tile 0 into buffer 0 ----
    for (int i = tid; i < TILE_B128; i += 256)
        async_copy_b128(protou + (size_t)i * 8, &sB[0][(size_t)i * 8]);

    for (int nt = 0; nt < NTILES; ++nt) {
        const int cur = nt & 1;

        wait_async0();     // my async writes into sB[cur] have landed
        __syncthreads();   // everyone's landed; everyone done reading sB[cur^1]

        // kick off DMA of the next tile into the other buffer
        if (nt + 1 < NTILES) {
            const unsigned short* g = protou + (size_t)(nt + 1) * NTILE * KP;
            for (int i = tid; i < TILE_B128; i += 256)
                async_copy_b128(g + (size_t)i * 8, &sB[cur ^ 1][(size_t)i * 8]);
        }

        // ---- compute on the current tile while the DMA flies ----
        const __bf16* sBbf = reinterpret_cast<const __bf16*>(sB[cur]);
        v8f acc = {};  // 16x16 fp32 cross-products for this wave's stripe
#pragma unroll
        for (int kt = 0; kt < KP; kt += 32) {
            // A fragment (16x32): lane M = nl,
            // K = rg*16 + {0..15}, contiguous
            v16bf afrag = *reinterpret_cast<const v16bf*>(aRow + kt + rg * 16);
            // B fragment (32x16): lane N = nl, K = rg*16 + {0..15}, contiguous
            v16bf bfrag = *reinterpret_cast<const v16bf*>(
                sBbf + (size_t)nl * KP + kt + rg * 16);
            acc = __builtin_amdgcn_wmma_f32_16x16x32_bf16(
                false, afrag, false, bfrag, (short)0, acc, false, false);
        }

        // fused RBF + linear-weight epilogue; C layout: VGPR r -> M=r+8*rg, N=nl
        const float pn = p2[nt * NTILE + nl];
        const float wn = lin_w[nt * NTILE + nl];
#pragma unroll
        for (int r = 0; r < 8; ++r) {
            float d2 = fmaxf(f2r[r] + pn - 2.0f * acc[r], 0.0f);
            rowsum[r] += __expf(-d2) * wn;   // GAMMA = 1
        }
    }

    // reduce the 16 N-lanes of each half-wave (xor masks stay within a half)
#pragma unroll
    for (int r = 0; r < 8; ++r) {
        float v = rowsum[r];
        for (int off = 1; off < 16; off <<= 1) v += __shfl_xor(v, off, 32);
        rowsum[r] = v;
    }

    if (nl == 0) {  // lanes 0 and 16 hold rows rg*8 + {0..7}
        const float bias = lin_b[0];  // SHIFT = 0
#pragma unroll
        for (int r = 0; r < 8; ++r) {
            const int row = m_base + rg * 8 + r;
            const float z  = rowsum[r] + bias;
            const float l  = 1.0f / (1.0f + __expf(-z));   // sigmoid
            const float p0 = l, p1 = 1.0f - l;
            const float mx  = fmaxf(p0, p1);
            const float lse = mx + logf(__expf(p0 - mx) + __expf(p1 - mx));
            out[2 * row + 0] = p0 - lse;
            out[2 * row + 1] = p1 - lse;
        }
    }
}

// ---------------------------------------------------------------------------
extern "C" void kernel_launch(void* const* d_in, const int* in_sizes, int n_in,
                              void* d_out, int out_size, void* d_ws,
                              size_t ws_size, hipStream_t stream) {
    (void)in_sizes; (void)n_in; (void)out_size; (void)ws_size;
    const float* x   = (const float*)d_in[0];  // (8192,1,28,28)
    const float* cw  = (const float*)d_in[1];  // (4,1,2,2)
    const float* cb  = (const float*)d_in[2];  // (4,)
    const float* pr  = (const float*)d_in[3];  // (2048,784)
    const float* lw  = (const float*)d_in[4];  // (1,2048)
    const float* lb  = (const float*)d_in[5];  // (1,)
    float* out = (float*)d_out;                // (8192,2)

    char* ws = (char*)d_ws;
    unsigned short* featsu = (unsigned short*)ws;                 // B*KP bf16
    size_t off = (size_t)BATCH * KP * sizeof(unsigned short);
    unsigned short* protou = (unsigned short*)(ws + off);         // P*KP bf16
    off += (size_t)PROTO * KP * sizeof(unsigned short);
    float* f2 = (float*)(ws + off);                               // B floats
    off += (size_t)BATCH * sizeof(float);
    float* p2 = (float*)(ws + off);                               // P floats

    quanv_conv_pack<<<BATCH, 256, 0, stream>>>(x, cw, cb, featsu, f2);
    proto_pack<<<PROTO, 256, 0, stream>>>(pr, protou, p2);
    rbf_gemm_head<<<BATCH / MTILE, 256, 0, stream>>>(featsu, protou, f2, p2,
                                                     lw, lb, out);
}
